// VirtualNode_38654705664489
// MI455X (gfx1250) — matrix operand order repaired and checked by reading, not compiled, not run on hardware
//
#include <hip/hip_runtime.h>

#define Nn 200000
#define Dd 256
#define Bb 512
#define LDSK 264  // padded LDS row stride (bf16 elems): 528B rows -> bank stride 4, conflict-free b128 reads

typedef __attribute__((ext_vector_type(8)))  float  v8f;
typedef __attribute__((ext_vector_type(16))) float  v16f;
typedef __attribute__((ext_vector_type(16))) __bf16 v16bf;

union BF16Vec {
  v16bf bf;
  unsigned short u[16];
  uint4 q[2];
};

__device__ __forceinline__ unsigned short f2bf(float f) {
  unsigned int u = __float_as_uint(f);
  u += 0x7FFFu + ((u >> 16) & 1u);   // round-to-nearest-even
  return (unsigned short)(u >> 16);
}

__device__ __forceinline__ float gelu_exact(float h) {
  return 0.5f * h * (1.0f + erff(h * 0.7071067811865475f));
}

// ---------------- K1: zero seg_sum + cnt (131584 floats contiguous) ----------------
__global__ __launch_bounds__(256) void k_zero(float* __restrict__ p) {
  int i = blockIdx.x * 256 + threadIdx.x;
  if (i < Bb * Dd + Bb) p[i] = 0.0f;
}

// ---------------- K2: scatter-mean partials (batch is sorted int64) ----------------
__global__ __launch_bounds__(256) void k_scatter(const float* __restrict__ x,
                                                 const long long* __restrict__ batch,
                                                 float* __restrict__ seg,
                                                 float* __restrict__ cntv) {
  __shared__ int sb[128];
  int t = threadIdx.x;
  long long base = (long long)blockIdx.x * 128;
  long long rem = (long long)Nn - base;
  int cnt_here = rem > 128 ? 128 : (int)rem;
  if (t < cnt_here) sb[t] = (int)batch[base + t];
  __syncthreads();
  int cur = sb[0];
  float run = 0.0f;
  for (int i = 0; i < cnt_here; ++i) {
    int b = sb[i];
    if (b != cur) {
      atomicAdd(&seg[(long long)cur * Dd + t], run);
      run = 0.0f;
      cur = b;
    }
    run += x[(base + i) * Dd + t];
  }
  atomicAdd(&seg[(long long)cur * Dd + t], run);
  if (t == 0) {
    cur = sb[0];
    int c = 0;
    for (int i = 0; i < cnt_here; ++i) {
      if (sb[i] != cur) { atomicAdd(&cntv[cur], (float)c); cur = sb[i]; c = 0; }
      ++c;
    }
    atomicAdd(&cntv[cur], (float)c);
  }
}

// ---------------- K3: per-segment fused MLPs: vn_state, U, V ----------------
__device__ __forceinline__ float2 block_sum2(float a, float b) {
  __shared__ float2 red[256];
  int t = threadIdx.x;
  red[t].x = a; red[t].y = b;
  __syncthreads();
  #pragma unroll
  for (int s = 128; s > 0; s >>= 1) {
    if (t < s) { red[t].x += red[t + s].x; red[t].y += red[t + s].y; }
    __syncthreads();
  }
  float2 r = red[0];
  __syncthreads();
  return r;
}

__global__ __launch_bounds__(256) void k_vn(const float* __restrict__ seg,
    const float* __restrict__ cntv, const float* __restrict__ vn_embed,
    const float* __restrict__ W1, const float* __restrict__ b1v,
    const float* __restrict__ g1, const float* __restrict__ be1,
    const float* __restrict__ W2, const float* __restrict__ b2v,
    const float* __restrict__ g2, const float* __restrict__ be2,
    const float* __restrict__ Wg,
    float* __restrict__ vn_state_out, float* __restrict__ U, float* __restrict__ V) {
  __shared__ float row[Dd];
  int r = blockIdx.x, t = threadIdx.x;
  float c = fmaxf(cntv[r], 1.0f);
  row[t] = seg[(long long)r * Dd + t] / c;
  __syncthreads();
  // h1 = gelu(agg @ W1 + b1)
  float h = b1v[t];
  #pragma unroll 8
  for (int k = 0; k < Dd; ++k) h += row[k] * W1[k * Dd + t];
  h = gelu_exact(h);
  float2 s = block_sum2(h, h * h);
  float mu = s.x * (1.0f / Dd);
  float var = s.y * (1.0f / Dd) - mu * mu;
  float vs = vn_embed[t] + (h - mu) * rsqrtf(var + 1e-5f) * g1[t] + be1[t];
  vn_state_out[(long long)r * Dd + t] = vs;
  __syncthreads();
  row[t] = vs;
  __syncthreads();
  // U = LN(gelu(vn_state @ W2 + b2))
  float h2 = b2v[t];
  #pragma unroll 8
  for (int k = 0; k < Dd; ++k) h2 += row[k] * W2[k * Dd + t];
  h2 = gelu_exact(h2);
  s = block_sum2(h2, h2 * h2);
  mu = s.x * (1.0f / Dd);
  var = s.y * (1.0f / Dd) - mu * mu;
  float uu = (h2 - mu) * rsqrtf(var + 1e-5f) * g2[t] + be2[t];
  U[(long long)r * Dd + t] = uu;
  __syncthreads();
  row[t] = uu;
  __syncthreads();
  // V = U @ Wg_bot   (Wg rows 256..511)
  float vv = 0.0f;
  #pragma unroll 8
  for (int k = 0; k < Dd; ++k) vv += row[k] * Wg[(Dd + k) * Dd + t];
  V[(long long)r * Dd + t] = vv;
}

// ---------------- K4: Wg_top -> bf16, transposed (WgTbf[n][k] = bf16(Wg[k][n])) ----------------
__global__ __launch_bounds__(256) void k_wcvt(const float* __restrict__ Wg,
                                              unsigned short* __restrict__ WgTbf) {
  int k = blockIdx.x, n = threadIdx.x;
  WgTbf[n * Dd + k] = f2bf(Wg[k * Dd + n]);
}

// ---------------- K5: WMMA gate GEMM (x @ Wg_top) + fused sigmoid gate + residual ----------------
__global__ __launch_bounds__(256) void k_gate(const float* __restrict__ x,
    const long long* __restrict__ batch, const unsigned short* __restrict__ WgTbf,
    const float* __restrict__ U, const float* __restrict__ V,
    const float* __restrict__ bg, float* __restrict__ x_out) {
  extern __shared__ unsigned short wg[];  // [256][LDSK] bf16, padded
  int t = threadIdx.x;
  // cooperative stage of Wg_top^T (bf16, 128KB payload) into LDS, 16B chunks
  for (int i = t; i < 256 * 32; i += 256) {
    int n = i >> 5, ch = i & 31;
    uint4 d = ((const uint4*)WgTbf)[n * 32 + ch];
    *((uint4*)&wg[n * LDSK + ch * 8]) = d;
  }
  __syncthreads();

  int wave = t >> 5, lane = t & 31;
  long long rbase = (long long)blockIdx.x * 128 + wave * 16;  // 16 rows per wave
  if (rbase >= Nn) return;                                    // wave-uniform, EXEC stays all-ones
  int m = lane & 15, q = lane >> 4;
  const float* xrow = x + (rbase + m) * Dd;

  v8f acc[16];
  #pragma unroll
  for (int i = 0; i < 16; ++i) acc[i] = (v8f)0.0f;

  for (int kc = 0; kc < 8; ++kc) {
    // A fragment 16x32 bf16: lane m=row, half-wave q selects K quarters {q*8..}, {16+q*8..}
    int k0 = kc * 32 + q * 8;
    float4 a0 = *(const float4*)(xrow + k0);
    float4 a1 = *(const float4*)(xrow + k0 + 4);
    float4 a2 = *(const float4*)(xrow + k0 + 16);
    float4 a3 = *(const float4*)(xrow + k0 + 20);
    v16f af = (v16f){a0.x, a0.y, a0.z, a0.w, a1.x, a1.y, a1.z, a1.w,
                     a2.x, a2.y, a2.z, a2.w, a3.x, a3.y, a3.z, a3.w};
    v16bf av = __builtin_convertvector(af, v16bf);  // RNE fptrunc -> v_cvt_pk_bf16_f32

    int khalf = kc * 32 + q * 8;  // B 32x16: lane col = m, half-wave q selects K quarters {q*8..}, {16+q*8..}
    #pragma unroll
    for (int nt = 0; nt < 16; ++nt) {
      int n = nt * 16 + m;
      const unsigned short* p = &wg[n * LDSK + khalf];
      BF16Vec bv;
      bv.q[0] = *(const uint4*)(p);      // 2x ds_load_b128, 16B aligned, conflict-free
      bv.q[1] = *(const uint4*)(p + 16);
      acc[nt] = __builtin_amdgcn_wmma_f32_16x16x32_bf16(
          false, av, false, bv.bf, (short)0, acc[nt], false, false);
    }
  }

  // epilogue: C/D layout lane m=N, M = v + 8q
  #pragma unroll
  for (int v = 0; v < 8; ++v) {
    long long row = rbase + q * 8 + v;
    long long bi = batch[row];
    const float* Ur = U + bi * Dd;
    const float* Vr = V + bi * Dd;
    const float* xr = x + row * Dd;
    float* outr = x_out + row * Dd;
    #pragma unroll
    for (int nt = 0; nt < 16; ++nt) {
      int col = nt * 16 + m;
      float pre = acc[nt][v] + Vr[col] + bg[col];
      float g = __builtin_amdgcn_rcpf(1.0f + __expf(-pre));  // v_rcp_f32 sigmoid
      outr[col] = xr[col] + g * Ur[col];
    }
  }
}

extern "C" void kernel_launch(void* const* d_in, const int* in_sizes, int n_in,
                              void* d_out, int out_size, void* d_ws, size_t ws_size,
                              hipStream_t stream) {
  const float* x        = (const float*)d_in[0];
  const long long* batch= (const long long*)d_in[1];
  const float* vn_embed = (const float*)d_in[2];
  const float* W1  = (const float*)d_in[3];
  const float* b1  = (const float*)d_in[4];
  const float* g1  = (const float*)d_in[5];
  const float* be1 = (const float*)d_in[6];
  const float* W2  = (const float*)d_in[7];
  const float* b2  = (const float*)d_in[8];
  const float* g2  = (const float*)d_in[9];
  const float* be2 = (const float*)d_in[10];
  const float* Wg  = (const float*)d_in[11];
  const float* bg  = (const float*)d_in[12];

  float* x_out = (float*)d_out;
  float* vn_state_out = x_out + (size_t)Nn * Dd;

  // workspace layout (floats): seg[131072] | cnt[512] | U[131072] | V[131072] | WgTbf(ushort 65536)
  float* ws_f = (float*)d_ws;
  float* seg  = ws_f;
  float* cntv = ws_f + 131072;
  float* U    = ws_f + 131584;
  float* V    = ws_f + 262656;
  unsigned short* WgTbf = (unsigned short*)(ws_f + 393728);

  (void)in_sizes; (void)n_in; (void)out_size; (void)ws_size;

  k_zero<<<(Bb * Dd + Bb + 255) / 256, 256, 0, stream>>>(seg);
  k_scatter<<<(Nn + 127) / 128, 256, 0, stream>>>(x, batch, seg, cntv);
  k_vn<<<Bb, 256, 0, stream>>>(seg, cntv, vn_embed, W1, b1, g1, be1,
                               W2, b2, g2, be2, Wg, vn_state_out, U, V);
  k_wcvt<<<Dd, 256, 0, stream>>>(Wg, WgTbf);
  k_gate<<<(Nn + 127) / 128, 256, 256 * LDSK * sizeof(unsigned short), stream>>>(
      x, batch, WgTbf, U, V, bg, x_out);
}
